// VariableSizeLSTM_85298050498918
// MI455X (gfx1250) — matrix operand order repaired and checked by reading, not compiled, not run on hardware
//
#include <hip/hip_runtime.h>

// ---------------------------------------------------------------------------
// 3-layer LSTM (H = 64,64,128, IN=32, B=4096, T=512) + final FC to 10.
// Strategy: one workgroup (4 wave32) per 16-row batch tile; all weights packed
// to f16 WMMA-B layout in LDS; h/x activation tiles in LDS; cell state c and
// gate accumulators in registers; v_wmma_f32_16x16x32_f16 for all GEMMs.
// ---------------------------------------------------------------------------

typedef __attribute__((ext_vector_type(16))) _Float16 v16h;
typedef __attribute__((ext_vector_type(8)))  _Float16 v8h;
typedef __attribute__((ext_vector_type(8)))  float    v8f;

#define LSTM_T   512
#define LSTM_IN  32
#define LSTM_BM  16
#define NTHREADS 128

// Packed-weight layout (units of _Float16):
//   per layer: [ntile][kb][lane(32)][elem(16)] ; ntile = gate*G + grp
#define W0_OFF 0          // KB=3 (K=96),  NT=16 -> 3*16*512  = 24576
#define W1_OFF 24576      // KB=4 (K=128), NT=16 -> 4*16*512  = 32768
#define W2_OFF 57344      // KB=6 (K=192), NT=32 -> 6*32*512  = 98304
#define WPACK_HALVES 155648
#define A0_OFF 155648     // 16 x 96   (x_t | h0_prev)
#define A1_OFF 157184     // 16 x 128  (h0    | h1_prev)
#define A2_OFF 159232     // 16 x 192  (h1    | h2_prev)
#define LDS_HALVES 162304
#define LDS_BYTES (LDS_HALVES * 2)   // 324608 B  (< 320 KB WGP limit)

// ---------------------------------------------------------------------------
// Prep kernel: f32 weights -> f16, packed into the WMMA B-operand lane layout.
// B (32x16, f16), contiguous k order (matches load_a_tile): lane = N%16 + 16*(K%32 >= 16), element j = K%16.
// K axis is the concatenation [x-input (Kx) | h-input (H)].
// ---------------------------------------------------------------------------
__global__ void lstm_pack_weights(const float* __restrict__ Wih0, const float* __restrict__ Whh0,
                                  const float* __restrict__ Wih1, const float* __restrict__ Whh1,
                                  const float* __restrict__ Wih2, const float* __restrict__ Whh2,
                                  _Float16* __restrict__ out)
{
  int idx = blockIdx.x * 256 + threadIdx.x;
  if (idx >= WPACK_HALVES) return;

  const float *Wih, *Whh;
  int rem, KB, Kx, H;
  if (idx < W1_OFF)      { rem = idx - W0_OFF; Wih = Wih0; Whh = Whh0; KB = 3; Kx = 32; H = 64;  }
  else if (idx < W2_OFF) { rem = idx - W1_OFF; Wih = Wih1; Whh = Whh1; KB = 4; Kx = 64; H = 64;  }
  else                   { rem = idx - W2_OFF; Wih = Wih2; Whh = Whh2; KB = 6; Kx = 64; H = 128; }

  int j     = rem & 15;
  int lane  = (rem >> 4) & 31;
  int kbnt  = rem >> 9;
  int kb    = kbnt % KB;
  int ntile = kbnt / KB;

  int N = ntile * 16 + (lane & 15);
  int K = kb * 32 + j + ((lane & 16) ? 16 : 0);

  float v = (K < Kx) ? Wih[N * Kx + K] : Whh[N * H + (K - Kx)];
  out[idx] = (_Float16)v;
}

// ---------------------------------------------------------------------------
// Device helpers
// ---------------------------------------------------------------------------
__device__ __forceinline__ float sigm(float x)      { return 1.0f / (1.0f + __expf(-x)); }
__device__ __forceinline__ float tanh_fast(float x) { float e = __expf(-2.0f * x); return (1.0f - e) / (1.0f + e); }

// A operand (16x32 f16) from row-major LDS tile.
// Lanes 0-15: elems 0-15 = K 0-15.
// Lanes 16-31: elems 0-15 = K 16-31.
__device__ __forceinline__ v16h load_a_tile(const _Float16* rowp, int lane)
{
  const int off = (lane & 16) ? 16 : 0;
  v8h lo = *(const v8h*)(rowp + off);
  v8h hi = *(const v8h*)(rowp + off + 8);
  v16h a;
#pragma unroll
  for (int j = 0; j < 8; ++j) { a[j] = lo[j]; a[j + 8] = hi[j]; }
  return a;
}

// Gates GEMM for one 16-column group: acc[gate] += A(16xK) * Wpacked(Kx16).
template <int KB, int G>
__device__ __forceinline__ void gemm_group(const _Float16* Abuf, int astride,
                                           const _Float16* Wsec, int grp, int lane,
                                           v8f acc[4])
{
  const _Float16* arow = Abuf + (lane & 15) * astride;
#pragma unroll
  for (int kb = 0; kb < KB; ++kb) {
    v16h a = load_a_tile(arow + kb * 32, lane);
#pragma unroll
    for (int gate = 0; gate < 4; ++gate) {
      v16h b = *(const v16h*)(Wsec + (((gate * G + grp) * KB + kb) * 32 + lane) * 16);
      acc[gate] = __builtin_amdgcn_wmma_f32_16x16x32_f16(
          false, a, false, b, (short)0, acc[gate], false, false);
    }
  }
}

// LSTM elementwise (fp32), cell state c in registers; write h (f16) to LDS.
// C/D layout: vgpr r -> row (lane<16 ? r : r+8), col = grp*16 + lane%16.
__device__ __forceinline__ void lstm_ew(const v8f acc[4], float4 b, v8f& c,
                                        int lane, int grp,
                                        _Float16* h1, int s1,
                                        _Float16* h2, int s2)
{
  const int col   = grp * 16 + (lane & 15);
  const int rbase = (lane & 16) ? 8 : 0;
#pragma unroll
  for (int r = 0; r < 8; ++r) {
    float ig = sigm(acc[0][r] + b.x);
    float fg = sigm(acc[1][r] + b.y);
    float gg = tanh_fast(acc[2][r] + b.z);
    float og = sigm(acc[3][r] + b.w);
    float cn = fg * c[r] + ig * gg;
    c[r] = cn;
    _Float16 h = (_Float16)(og * tanh_fast(cn));
    h1[(rbase + r) * s1 + col] = h;
    if (h2) h2[(rbase + r) * s2 + col] = h;
  }
}

// ---------------------------------------------------------------------------
// Main persistent kernel: 256 blocks x 128 threads (4 wave32), 16 batch rows.
// ---------------------------------------------------------------------------
__global__ __launch_bounds__(NTHREADS)
void lstm_main_kernel(const float* __restrict__ x,
                      const _Float16* __restrict__ wpack,
                      const float* __restrict__ bih0, const float* __restrict__ bhh0,
                      const float* __restrict__ bih1, const float* __restrict__ bhh1,
                      const float* __restrict__ bih2, const float* __restrict__ bhh2,
                      const float* __restrict__ Wfc,  const float* __restrict__ bfc,
                      float* __restrict__ out)
{
  extern __shared__ float4 smem4[];           // 16B-aligned LDS base
  _Float16* lds = (_Float16*)smem4;
  _Float16* W0 = lds + W0_OFF;
  _Float16* W1 = lds + W1_OFF;
  _Float16* W2 = lds + W2_OFF;
  _Float16* A0 = lds + A0_OFF;                // 16 x 96
  _Float16* A1 = lds + A1_OFF;                // 16 x 128
  _Float16* A2 = lds + A2_OFF;                // 16 x 192

  const int tid  = threadIdx.x;
  const int lane = tid & 31;
  const int w    = tid >> 5;                  // wave id 0..3 = column group
  const int row0 = blockIdx.x * LSTM_BM;
  const int cw   = lane & 15;

  // Stage all packed weights into LDS once (19456 float4), zero A buffers.
  {
    const float4* src = (const float4*)wpack;
    for (int i = tid; i < (WPACK_HALVES / 8); i += NTHREADS) smem4[i] = src[i];
    for (int i = (WPACK_HALVES / 8) + tid; i < (LDS_HALVES / 8); i += NTHREADS)
      smem4[i] = make_float4(0.f, 0.f, 0.f, 0.f);
  }

  // Per-lane combined biases (bih+bhh), one float per gate, per column group.
  float4 b0, b1, b2a, b2b;
  {
    int n = w * 16 + cw;                                        // H=64 layers
    b0 = make_float4(bih0[n] + bhh0[n], bih0[64 + n] + bhh0[64 + n],
                     bih0[128 + n] + bhh0[128 + n], bih0[192 + n] + bhh0[192 + n]);
    b1 = make_float4(bih1[n] + bhh1[n], bih1[64 + n] + bhh1[64 + n],
                     bih1[128 + n] + bhh1[128 + n], bih1[192 + n] + bhh1[192 + n]);
    int na = (2 * w) * 16 + cw, nb = (2 * w + 1) * 16 + cw;     // H=128 layer
    b2a = make_float4(bih2[na] + bhh2[na], bih2[128 + na] + bhh2[128 + na],
                      bih2[256 + na] + bhh2[256 + na], bih2[384 + na] + bhh2[384 + na]);
    b2b = make_float4(bih2[nb] + bhh2[nb], bih2[128 + nb] + bhh2[128 + nb],
                      bih2[256 + nb] + bhh2[256 + nb], bih2[384 + nb] + bhh2[384 + nb]);
  }

  const v8f vz = {0.f, 0.f, 0.f, 0.f, 0.f, 0.f, 0.f, 0.f};
  v8f c0 = vz, c1 = vz, c2a = vz, c2b = vz;   // cell states live in registers

  __syncthreads();

  for (int t = 0; t < LSTM_T; ++t) {
    // ---- stage x_t (f32 -> f16) into A0 cols [0,32) ----
    {
      int r  = tid >> 3;                       // 0..15
      int k4 = (tid & 7) << 2;                 // 0,4,...,28
      const float* xp = x + ((size_t)(row0 + r) * LSTM_T + t) * LSTM_IN + k4;
      float4 xv = *(const float4*)xp;
      if (t + 1 < LSTM_T) __builtin_prefetch(xp + LSTM_IN, 0, 0);  // next step
      _Float16* d = A0 + r * 96 + k4;
      d[0] = (_Float16)xv.x; d[1] = (_Float16)xv.y;
      d[2] = (_Float16)xv.z; d[3] = (_Float16)xv.w;
    }
    __syncthreads();

    // ---- layer 0: A = [x_t | h0_prev] (K=96), 4H=256, group = w ----
    {
      v8f acc[4] = {vz, vz, vz, vz};
      gemm_group<3, 4>(A0, 96, W0, w, lane, acc);
      __syncthreads();                          // all reads of A0 done
      lstm_ew(acc, b0, c0, lane, w, A0 + 32, 96, A1, 128);  // h0 -> A0, A1
    }
    __syncthreads();

    // ---- layer 1: A = [h0 | h1_prev] (K=128), 4H=256, group = w ----
    {
      v8f acc[4] = {vz, vz, vz, vz};
      gemm_group<4, 4>(A1, 128, W1, w, lane, acc);
      __syncthreads();
      lstm_ew(acc, b1, c1, lane, w, A1 + 64, 128, A2, 192); // h1 -> A1, A2
    }
    __syncthreads();

    // ---- layer 2: A = [h1 | h2_prev] (K=192), 4H=512, groups 2w, 2w+1 ----
    {
      v8f accA[4] = {vz, vz, vz, vz};
      v8f accB[4] = {vz, vz, vz, vz};
      gemm_group<6, 8>(A2, 192, W2, 2 * w,     lane, accA);
      gemm_group<6, 8>(A2, 192, W2, 2 * w + 1, lane, accB);
      __syncthreads();
      lstm_ew(accA, b2a, c2a, lane, 2 * w,     A2 + 64, 192, (_Float16*)nullptr, 0);
      lstm_ew(accB, b2b, c2b, lane, 2 * w + 1, A2 + 64, 192, (_Float16*)nullptr, 0);
    }
    __syncthreads();
  }

  // ---- final FC: out[16,10] = h2 @ Wfc^T + bfc (tiny; plain VALU) ----
  for (int idx = tid; idx < LSTM_BM * 10; idx += NTHREADS) {
    int r = idx / 10, o = idx % 10;
    float s = bfc[o];
    const _Float16* hp = A2 + r * 192 + 64;   // final h2 row (f16)
    for (int k = 0; k < 128; ++k) s += (float)hp[k] * Wfc[o * 128 + k];
    out[(size_t)(row0 + r) * 10 + o] = s;
  }
}

// ---------------------------------------------------------------------------
extern "C" void kernel_launch(void* const* d_in, const int* in_sizes, int n_in,
                              void* d_out, int out_size, void* d_ws, size_t ws_size,
                              hipStream_t stream)
{
  (void)in_sizes; (void)n_in; (void)out_size; (void)ws_size;

  const float* x    = (const float*)d_in[0];
  const float* Wih0 = (const float*)d_in[1];
  const float* Whh0 = (const float*)d_in[2];
  const float* bih0 = (const float*)d_in[3];
  const float* bhh0 = (const float*)d_in[4];
  const float* Wih1 = (const float*)d_in[5];
  const float* Whh1 = (const float*)d_in[6];
  const float* bih1 = (const float*)d_in[7];
  const float* bhh1 = (const float*)d_in[8];
  const float* Wih2 = (const float*)d_in[9];
  const float* Whh2 = (const float*)d_in[10];
  const float* bih2 = (const float*)d_in[11];
  const float* bhh2 = (const float*)d_in[12];
  const float* Wfc  = (const float*)d_in[13];
  const float* bfc  = (const float*)d_in[14];

  _Float16* wpack = (_Float16*)d_ws;          // 311296 B of scratch
  float*    out   = (float*)d_out;

  // Allow 324,608 B of dynamic LDS (< 320 KB per-workgroup limit).
  hipFuncSetAttribute((const void*)lstm_main_kernel,
                      hipFuncAttributeMaxDynamicSharedMemorySize, LDS_BYTES);

  // 1) pack f32 weights -> f16 WMMA-B layout in workspace
  lstm_pack_weights<<<(WPACK_HALVES + 255) / 256, 256, 0, stream>>>(
      Wih0, Whh0, Wih1, Whh1, Wih2, Whh2, wpack);

  // 2) persistent LSTM: 4096/16 = 256 workgroups, 128 threads each
  lstm_main_kernel<<<dim3(4096 / LSTM_BM), dim3(NTHREADS), LDS_BYTES, stream>>>(
      x, wpack, bih0, bhh0, bih1, bhh1, bih2, bhh2, Wfc, bfc, out);
}
